// MultiHeadAttention_7103875908219
// MI455X (gfx1250) — matrix operand order repaired and checked by reading, not compiled, not run on hardware
//
#include <hip/hip_runtime.h>

// ---------------------------------------------------------------------------
// MI455X (gfx1250) attention layer:
//   Q/K/V projections (pipelined WMMA bf16 GEMM) -> RoPE -> flash attention
//   (TDM tensor_load_to_lds + async-to-LDS staging, WMMA bf16, tanh soft-cap,
//   causal, online softmax) -> output projection.
// ---------------------------------------------------------------------------

typedef __attribute__((ext_vector_type(16))) __bf16 v16bf;
typedef __attribute__((ext_vector_type(8)))  float  v8f;
typedef __attribute__((ext_vector_type(4)))  unsigned u32x4;
typedef __attribute__((ext_vector_type(4)))  int      i32x4;
typedef __attribute__((ext_vector_type(8)))  int      i32x8;

struct F8 { union { v8f v; float f[8]; }; };

// Native f32 -> bf16 (hardware cvt, RNE).
__device__ __forceinline__ unsigned short f2bf(float x) {
    union { __bf16 h; unsigned short u; } c;
    c.h = (__bf16)x;
    return c.u;
}
// Packed pair: one v_cvt_pk_bf16_f32 (two f32 -> two bf16 in one VGPR).
__device__ __forceinline__ unsigned pk2bf(float lo, float hi) {
    unsigned r;
    asm("v_cvt_pk_bf16_f32 %0, %1, %2" : "=v"(r) : "v"(lo), "v"(hi));
    return r;
}

// Load a 16-element bf16 fragment as two 16B LDS chunks.
__device__ __forceinline__ v16bf frag16(const unsigned short* p0,
                                        const unsigned short* p1) {
    union { uint4 q[2]; v16bf v; } u;
    u.q[0] = *(const uint4*)p0;
    u.q[1] = *(const uint4*)p1;
    return u.v;
}

__device__ __forceinline__ v8f wmma_bf16(v16bf a, v16bf b, v8f c) {
    return __builtin_amdgcn_wmma_f32_16x16x32_bf16(
        false, a, false, b, (short)0, c, false, false);
}

// Async DMA: copy 16 bytes global -> LDS, tracked by ASYNCcnt (CDNA5).
__device__ __forceinline__ void async_copy_b128(void* lds_ptr,
                                                const void* gptr) {
    unsigned           ldsa = (unsigned)(unsigned long long)lds_ptr;
    unsigned long long ga   = (unsigned long long)gptr;
    asm volatile("global_load_async_to_lds_b128 %0, %1, off"
                 :: "v"(ldsa), "v"(ga) : "memory");
}
__device__ __forceinline__ void wait_asynccnt0() {
    asm volatile("s_wait_asynccnt 0x0" ::: "memory");
}

// ---------------------------------------------------------------------------
// TDM: load a 2D bf16 tile (rows x rowlen elems, global row stride in elems)
// into LDS, padding each LDS row by 16B (4 DWORDs). pad_iv encodes the data
// DWORDs per row: 5 -> 64 DW (128 elems), 4 -> 32 DW (64 elems).
// Issued by one wave; tracked by TENSORcnt.
// ---------------------------------------------------------------------------
#if __has_builtin(__builtin_amdgcn_tensor_load_to_lds)
#define HAVE_TDM 1
__device__ __forceinline__ void tdm_load_2d(void* lds, const void* gsrc,
                                            unsigned rowlen, unsigned rows,
                                            unsigned long long gstride,
                                            unsigned pad_iv) {
    unsigned long long ga = (unsigned long long)gsrc;
    u32x4 g0;
    g0[0] = 1u;                                          // count=1, user mode
    g0[1] = (unsigned)(unsigned long long)lds;           // lds_addr (bytes)
    g0[2] = (unsigned)(ga & 0xffffffffull);              // global_addr[31:0]
    g0[3] = (unsigned)((ga >> 32) & 0x1ffffffull)        // global_addr[56:32]
          | (2u << 30);                                  // type=2 (image)
    i32x8 g1;
    g1[0] = (int)((1u << 16)                              // data_size=2B
                | (1u << 20)                              // pad_enable
                | (pad_iv << 22)                          // pad_interval
                | (3u << 25));                            // pad_amount=4 DW
    g1[1] = (int)((rowlen & 0xffffu) << 16);              // tensor_dim0 lo16
    g1[2] = (int)(((rowlen >> 16) & 0xffffu)              // tensor_dim0 hi16
                | ((rows & 0xffffu) << 16));              // tensor_dim1 lo16
    g1[3] = (int)(((rows >> 16) & 0xffffu)                // tensor_dim1 hi16
                | ((rowlen & 0xffffu) << 16));            // tile_dim0
    g1[4] = (int)(rows & 0xffffu);                        // tile_dim1 (tile_dim2=0)
    g1[5] = (int)(gstride & 0xffffffffull);               // dim0_stride[31:0]
    g1[6] = (int)((gstride >> 32) & 0xffffull);           // dim0_stride[47:32]
    g1[7] = 0;                                            // dim1_stride hi
    i32x4 z4 = {0, 0, 0, 0};
#if __clang_major__ >= 23
    i32x8 z8 = {0, 0, 0, 0, 0, 0, 0, 0};
    __builtin_amdgcn_tensor_load_to_lds(g0, g1, z4, z4, z8, 0);
#else
    __builtin_amdgcn_tensor_load_to_lds(g0, g1, z4, z4, 0);
#endif
}
#else
#define HAVE_TDM 0
#endif

// Clamped fast tanh via expf (tanh saturates for |x| > 15).
__device__ __forceinline__ float fast_tanh(float x) {
    x = fminf(fmaxf(x, -15.0f), 15.0f);
    float e = __expf(2.0f * x);
    return (e - 1.0f) / (e + 1.0f);
}

// ---------------------------------------------------------------------------
// Pipelined GEMM: C[M,N] = A[M,K] * B[K,N]; fp32 in, bf16 WMMA, fp32 out.
// Block tile 128x128, 8 waves (2x4), wave tile 64x32, K-step 32.
// Global loads for step k+32 are issued while WMMAs for step k run.
// ---------------------------------------------------------------------------
__global__ __launch_bounds__(256)
void gemm_bf16wmma(const float* __restrict__ A, const float* __restrict__ Bm,
                   float* __restrict__ C, int M, int N, int K) {
    const int tid  = threadIdx.x;
    const int w    = tid >> 5;
    const int lane = tid & 31;
    const int hi   = lane >> 4;
    const int l15  = lane & 15;
    const int wm   = w >> 2;     // 0..1
    const int wn   = w & 3;      // 0..3
    const long m0  = (long)blockIdx.y * 128;
    const long n0  = (long)blockIdx.x * 128;

    __shared__ unsigned short sA[128 * 40];   // [m][k], stride 40 (padded)
    __shared__ unsigned short sB[128 * 40];   // [n][k] (B transposed), stride 40

    F8 acc[4][2];
    for (int mi = 0; mi < 4; ++mi)
        for (int ni = 0; ni < 2; ++ni)
            for (int i = 0; i < 8; ++i) acc[mi][ni].f[i] = 0.0f;

    float4 ra[4], rb[4];
    // ---- prologue: load first K-tile into registers ----
    for (int s = 0; s < 4; ++s) {
        int slot = tid + s * 256;
        int r = slot >> 3, q = slot & 7;
        ra[s] = *(const float4*)(A + (m0 + r) * (long)K + q * 4);
        int kr = slot >> 5, nq = slot & 31;
        rb[s] = *(const float4*)(Bm + (long)kr * N + n0 + nq * 4);
    }

    for (int k0 = 0; k0 < K; k0 += 32) {
        // ---- registers -> LDS (fp32 -> bf16, packed hardware cvt) ----
        for (int s = 0; s < 4; ++s) {
            int slot = tid + s * 256;
            int r = slot >> 3, q = slot & 7;
            uint2 p;
            p.x = pk2bf(ra[s].x, ra[s].y);
            p.y = pk2bf(ra[s].z, ra[s].w);
            *(uint2*)(sA + r * 40 + q * 4) = p;
            int kr = slot >> 5, nq = slot & 31;
            sB[(nq * 4 + 0) * 40 + kr] = f2bf(rb[s].x);
            sB[(nq * 4 + 1) * 40 + kr] = f2bf(rb[s].y);
            sB[(nq * 4 + 2) * 40 + kr] = f2bf(rb[s].z);
            sB[(nq * 4 + 3) * 40 + kr] = f2bf(rb[s].w);
        }
        __syncthreads();

        // ---- issue next tile's global loads (hidden under WMMAs) ----
        const int kn = k0 + 32;
        if (kn < K) {
            for (int s = 0; s < 4; ++s) {
                int slot = tid + s * 256;
                int r = slot >> 3, q = slot & 7;
                ra[s] = *(const float4*)(A + (m0 + r) * (long)K + kn + q * 4);
                int kr = slot >> 5, nq = slot & 31;
                rb[s] = *(const float4*)(Bm + (long)(kn + kr) * N + n0 + nq * 4);
            }
            if (kn + 32 < K) {   // L2 prefetch two steps ahead
                int r = tid >> 3;
                __builtin_prefetch(A + (m0 + r) * (long)K + kn + 32, 0, 1);
                int kr = tid >> 5;
                __builtin_prefetch(Bm + (long)(kn + 32 + kr) * N + n0, 0, 1);
            }
        }

        // ---- WMMA over the staged tile ----
        v16bf af[4], bfr[2];
        for (int mi = 0; mi < 4; ++mi) {
            const unsigned short* p =
                sA + (wm * 64 + mi * 16 + l15) * 40 + hi * 8;
            af[mi] = frag16(p, p + 16);
        }
        for (int ni = 0; ni < 2; ++ni) {
            const unsigned short* p =
                sB + (wn * 32 + ni * 16 + l15) * 40 + hi * 8;
            bfr[ni] = frag16(p, p + 16);
        }
        for (int mi = 0; mi < 4; ++mi)
            for (int ni = 0; ni < 2; ++ni)
                acc[mi][ni].v = wmma_bf16(af[mi], bfr[ni], acc[mi][ni].v);
        __syncthreads();
    }

    // ---- epilogue: C layout: VGPR i -> (M = base+i+8*hi, N = l15) ----
    for (int mi = 0; mi < 4; ++mi)
        for (int ni = 0; ni < 2; ++ni) {
            long col  = n0 + wn * 32 + ni * 16 + l15;
            long rowb = m0 + wm * 64 + mi * 16 + hi * 8;
            for (int i = 0; i < 8; ++i)
                C[(rowb + i) * (long)N + col] = acc[mi][ni].f[i];
        }
}

// ---------------------------------------------------------------------------
// RoPE + cast to bf16: in (B,T,H,128) fp32 -> out bf16.
// ---------------------------------------------------------------------------
__global__ __launch_bounds__(256)
void rope_cast_kernel(const float* __restrict__ in,
                      unsigned short* __restrict__ out, int T, int H) {
    long idx = (long)blockIdx.x * blockDim.x + threadIdx.x;  // B*T*H*64
    int  d   = (int)(idx & 63);
    long rh  = idx >> 6;                 // (b*T + t)*H + h
    int  t   = (int)((rh / H) % T);
    long base = rh * 128;

    float x1 = in[base + d];
    float x2 = in[base + d + 64];
    float inv = __expf(-9.210340371976184f * (float)(2 * d) * (1.0f / 128.0f));
    float ph  = (float)t * inv;
    float sn, cs;
    __sincosf(ph, &sn, &cs);
    out[base + d]      = f2bf(x1 * cs - x2 * sn);
    out[base + d + 64] = f2bf(x2 * cs + x1 * sn);
}

// ---------------------------------------------------------------------------
// V: fp32 (b,t,h,dh) -> bf16 transposed (b,h,dh,t) so the attention kernel
// can DMA V^T tiles straight into LDS in WMMA B-operand layout.
// ---------------------------------------------------------------------------
__global__ __launch_bounds__(256)
void vtrans_cast_kernel(const float* __restrict__ in,
                        unsigned short* __restrict__ out,
                        int T, int HKV) {
    long idx = (long)blockIdx.x * blockDim.x + threadIdx.x;  // B*T*HKV*128
    int  d   = (int)(idx & 127);
    long r   = idx >> 7;                  // (b*T + t)*HKV + h
    int  h   = (int)(r % HKV);
    long bt  = r / HKV;                   // b*T + t
    int  t   = (int)(bt % T);
    int  b   = (int)(bt / T);
    out[((long)((b * HKV + h) * 128 + d)) * T + t] = f2bf(in[idx]);
}

// ---------------------------------------------------------------------------
// Flash attention with tanh soft-cap + causal mask + GQA (g = HQ/HKV).
// Grid: (T/64, HQ, B). Block: 128 threads (4 waves); 16 query rows per wave.
// K/V tiles staged by the Tensor Data Mover (TENSORcnt); Q tile by
// GLOBAL_LOAD_ASYNC_TO_LDS_B128 (ASYNCcnt).
// ---------------------------------------------------------------------------
#define ATTN_MULT 0.08838834764831845f
#define MAX_ATTN  30.0f

__global__ __launch_bounds__(128)
void attn_kernel(const unsigned short* __restrict__ Qb,
                 const unsigned short* __restrict__ Kb,
                 const unsigned short* __restrict__ VbT,
                 float* __restrict__ Ctx, int T, int HQ, int HKV) {
    const int tid  = threadIdx.x;
    const int w    = tid >> 5;
    const int lane = tid & 31;
    const int hi   = lane >> 4;
    const int l15  = lane & 15;
    const int qt0  = blockIdx.x * 64;
    const int hq   = blockIdx.y;
    const int b    = blockIdx.z;
    const int hk   = hq / (HQ / HKV);

    __shared__ unsigned short sQ[64 * 136];   // [qrow][dh]
    __shared__ unsigned short sK[64 * 136];   // [key][dh]   == B-layout [n][k]
    __shared__ unsigned short sV[128 * 72];   // [dh][key]   == B-layout [n][k]
    __shared__ unsigned short sP[64 * 72];    // per-wave P rows [qrow][key]

    // ---- async-load Q tile (64 x 128 bf16) ----
    for (int s = 0; s < 8; ++s) {
        int slot = tid + s * 128;             // 1024 uint4 slots
        int r  = slot >> 4;
        int q4 = slot & 15;
        long g = ((long)(b * (long)T + qt0 + r) * HQ + hq) * 128 + q4 * 8;
        async_copy_b128(sQ + r * 136 + q4 * 8, Qb + g);
    }
    wait_asynccnt0();
    __syncthreads();

    v16bf aq[4];
    for (int kk = 0; kk < 4; ++kk) {
        const unsigned short* p = sQ + (w * 16 + l15) * 136 + kk * 32 + hi * 8;
        aq[kk] = frag16(p, p + 16);
    }

    F8 o[8];
    float mrow[8], lrow[8];
    for (int jj = 0; jj < 8; ++jj)
        for (int i = 0; i < 8; ++i) o[jj].f[i] = 0.0f;
    for (int i = 0; i < 8; ++i) { mrow[i] = -1e30f; lrow[i] = 0.0f; }

    const int nkb = blockIdx.x + 1;           // causal: key blocks <= query block
    for (int kb = 0; kb < nkb; ++kb) {
        __syncthreads();                      // retire reads of previous sK/sV
        // ---- stage K tile [key][dh] and V^T tile [dh][key] ----
#if HAVE_TDM
        if (w == 0) {
            long gk = ((long)(b * (long)T + kb * 64) * HKV + hk) * 128;
            tdm_load_2d(sK, Kb + gk, 128, 64,
                        (unsigned long long)(HKV * 128), 5 /* 64 DW rows */);
            long gv = ((long)((b * HKV + hk) * 128)) * T + kb * 64;
            tdm_load_2d(sV, VbT + gv, 64, 128,
                        (unsigned long long)T, 4 /* 32 DW rows */);
            __builtin_amdgcn_s_wait_tensorcnt((short)0);
        }
#else
        for (int s = 0; s < 8; ++s) {
            int slot = tid + s * 128;
            int r  = slot >> 4;
            int q4 = slot & 15;
            long g = ((long)(b * (long)T + kb * 64 + r) * HKV + hk) * 128 + q4 * 8;
            async_copy_b128(sK + r * 136 + q4 * 8, Kb + g);
            int dh = slot >> 3;
            int kc = slot & 7;
            long gv = ((long)((b * HKV + hk) * 128 + dh)) * T + kb * 64 + kc * 8;
            async_copy_b128(sV + dh * 72 + kc * 8, VbT + gv);
        }
        wait_asynccnt0();
#endif
        __syncthreads();

        // ---- S = Q * K^T : 16 (rows) x 64 (keys) per wave ----
        F8 sacc[4];
        for (int j = 0; j < 4; ++j)
            for (int i = 0; i < 8; ++i) sacc[j].f[i] = 0.0f;
        for (int j = 0; j < 4; ++j)
            for (int kk = 0; kk < 4; ++kk) {
                const unsigned short* p =
                    sK + (j * 16 + l15) * 136 + kk * 32 + hi * 8;
                v16bf bk = frag16(p, p + 16);
                sacc[j].v = wmma_bf16(aq[kk], bk, sacc[j].v);
            }

        // ---- soft-cap (+ causal mask on diagonal block), online softmax ----
        const bool diag = (kb == blockIdx.x);
        float mnew[8];
        for (int i = 0; i < 8; ++i) mnew[i] = mrow[i];
        for (int j = 0; j < 4; ++j) {
            int col = kb * 64 + j * 16 + l15;
            for (int i = 0; i < 8; ++i) {
                float s = sacc[j].f[i] * ATTN_MULT;
                s = MAX_ATTN * fast_tanh(s * (1.0f / MAX_ATTN));
                if (diag) {
                    int row = qt0 + w * 16 + i + hi * 8;
                    s = (col <= row) ? s : -1e30f;
                }
                sacc[j].f[i] = s;
                mnew[i] = fmaxf(mnew[i], s);
            }
        }
        for (int i = 0; i < 8; ++i)
            for (int off = 1; off < 16; off <<= 1)
                mnew[i] = fmaxf(mnew[i], __shfl_xor(mnew[i], off, 32));

        float scale[8], rowsum[8];
        for (int i = 0; i < 8; ++i) {
            scale[i]  = __expf(mrow[i] - mnew[i]);
            rowsum[i] = 0.0f;
        }
        for (int j = 0; j < 4; ++j)
            for (int i = 0; i < 8; ++i) {
                float p = __expf(sacc[j].f[i] - mnew[i]);
                rowsum[i] += p;
                sP[(w * 16 + i + hi * 8) * 72 + j * 16 + l15] = f2bf(p);
            }
        for (int i = 0; i < 8; ++i)
            for (int off = 1; off < 16; off <<= 1)
                rowsum[i] += __shfl_xor(rowsum[i], off, 32);
        for (int i = 0; i < 8; ++i) {
            lrow[i] = lrow[i] * scale[i] + rowsum[i];
            mrow[i] = mnew[i];
        }
        for (int jj = 0; jj < 8; ++jj)
            for (int i = 0; i < 8; ++i) o[jj].f[i] *= scale[i];

        // ---- O += P * V ----
        for (int kk2 = 0; kk2 < 2; ++kk2) {
            const unsigned short* pa =
                sP + (w * 16 + l15) * 72 + kk2 * 32 + hi * 8;
            v16bf ap = frag16(pa, pa + 16);
            for (int jj = 0; jj < 8; ++jj) {
                const unsigned short* pb =
                    sV + (jj * 16 + l15) * 72 + kk2 * 32 + hi * 8;
                v16bf bv = frag16(pb, pb + 16);
                o[jj].v = wmma_bf16(ap, bv, o[jj].v);
            }
        }
    }

    // ---- epilogue: context fp32 ----
    for (int jj = 0; jj < 8; ++jj) {
        int col = jj * 16 + l15;
        for (int i = 0; i < 8; ++i) {
            int row = qt0 + w * 16 + i + hi * 8;
            Ctx[((long)(b * (long)T + row) * HQ + hq) * 128 + col] =
                o[jj].f[i] / lrow[i];
        }
    }
}

// ---------------------------------------------------------------------------
// Host launcher
// ---------------------------------------------------------------------------
extern "C" void kernel_launch(void* const* d_in, const int* in_sizes, int n_in,
                              void* d_out, int out_size, void* d_ws,
                              size_t ws_size, hipStream_t stream) {
    const float* Xq = (const float*)d_in[0];
    const float* Xk = (const float*)d_in[1];
    const float* Xv = (const float*)d_in[2];
    // d_in[3] = mask (unused; causal mask is computed analytically)
    const float* Wq = (const float*)d_in[4];
    const float* Wk = (const float*)d_in[5];
    const float* Wv = (const float*)d_in[6];
    const float* Wo = (const float*)d_in[7];
    float* out = (float*)d_out;

    const int  B = 2, T = 2048, D = 4096, HQ = 32, HKV = 8;
    const long BT = (long)B * T;   // 4096

    char* ws = (char*)d_ws;
    float*          Qf  = (float*)(ws + 0x0000000);   // 67 MB (reused as Ctx)
    float*          Kf  = (float*)(ws + 0x4000000);   // 16 MB
    float*          Vf  = (float*)(ws + 0x5000000);   // 16 MB
    unsigned short* Qb  = (unsigned short*)(ws + 0x6000000);  // 32 MB
    unsigned short* Kb  = (unsigned short*)(ws + 0x8000000);  //  8 MB
    unsigned short* VbT = (unsigned short*)(ws + 0x8800000);  //  8 MB
    float*          Ctx = Qf;                         // reuse after RoPE

    // 1) projections (fp32 -> WMMA bf16 -> fp32)
    gemm_bf16wmma<<<dim3(4096 / 128, BT / 128), 256, 0, stream>>>(
        Xq, Wq, Qf, (int)BT, 4096, D);
    gemm_bf16wmma<<<dim3(1024 / 128, BT / 128), 256, 0, stream>>>(
        Xk, Wk, Kf, (int)BT, 1024, D);
    gemm_bf16wmma<<<dim3(1024 / 128, BT / 128), 256, 0, stream>>>(
        Xv, Wv, Vf, (int)BT, 1024, D);

    // 2) RoPE + casts (V also transposed for DMA-friendly attention staging)
    rope_cast_kernel<<<(unsigned)((BT * HQ * 64) / 256), 256, 0, stream>>>(
        Qf, Qb, T, HQ);
    rope_cast_kernel<<<(unsigned)((BT * HKV * 64) / 256), 256, 0, stream>>>(
        Kf, Kb, T, HKV);
    vtrans_cast_kernel<<<(unsigned)((BT * HKV * 128) / 256), 256, 0, stream>>>(
        Vf, VbT, T, HKV);

    // 3) flash attention (causal, tanh cap, GQA, TDM + async staging)
    attn_kernel<<<dim3(T / 64, HQ, B), 128, 0, stream>>>(
        Qb, Kb, VbT, Ctx, T, HQ, HKV);

    // 4) output projection
    gemm_bf16wmma<<<dim3(4096 / 128, BT / 128), 256, 0, stream>>>(
        Ctx, Wo, out, (int)BT, D, 4096);
}
